// CausalSelfAttention_11158325035217
// MI455X (gfx1250) — compile-verified, matrix operand order checked by reading, not run on hardware
//
#include <hip/hip_runtime.h>

// Chunked IPLR (DeltaNet-style) scan for B=4,H=16,T=4096,N=64,CHUNK=256 on gfx1250.
//
// Math restructure (see analysis): with w = u + v the per-chunk update becomes a
// block-16 recurrence  t = v_i + a_i@S ; w_i = (I - tril(a_i b_i^T,-1))^{-1} t ;
// S += b_i^T @ w_i,  and y_chunk = q_chunk @ S_end. One workgroup per (b,h),
// 8 waves; S lives in VGPR accumulators (16 WMMA C-tiles, 2 per wave) with a
// bf16 mirror in LDS (transposed, as the WMMA B-operand for a@S and q@S).

#define TLEN   4096
#define NDIM   64
#define CHUNKL 256
#define NCHUNK 16

typedef __bf16 bf16_t;
typedef bf16_t v16bf __attribute__((ext_vector_type(16)));
typedef bf16_t v8bf  __attribute__((ext_vector_type(8)));
typedef bf16_t v4bf  __attribute__((ext_vector_type(4)));
typedef float  v8f   __attribute__((ext_vector_type(8)));

// LDS pitches (elements). All row strides are 16-byte multiples so operand
// loads can be 128-bit ds loads; odd-dword strides where it matters for banks.
#define PA  72   // sA   : a = -k, bf16 [256][72]
#define PS  72   // sST  : S^T bf16 [64][72]   (sST[q][p] = S[p][q])
#define PB  32   // sBT/sWT : bf16 [64][32], K padded 16->32 with zeros
#define PT  65   // sT   : f32 [16][65]
#define PAB 17   // sAB  : f32 [16][17]

// ---- WMMA operand loaders (ISA 7.12.2 layouts) ----------------------------
// A (16x32 bf16): lane l<16 -> row M=l, K = {k0..k0+7, k0+16..k0+23};
//                 lane l>=16 -> row M=l-16, K = {k0+8..k0+15, k0+24..k0+31}.
// B uses the identical striping with N in place of M, so a B operand is read
// from an [N][K] (i.e. transposed) array with the same helper.
__device__ __forceinline__ v16bf load_tile_lds(const bf16_t* src, int rowBase,
                                               int pitch, int k0, int lane) {
  const int half = lane >> 4, l = lane & 15;
  const bf16_t* p = src + (rowBase + l) * pitch + k0 + half * 8;
  v8bf lo = *(const v8bf*)p;          // 16B aligned: pitch%8==0, k0%8==0
  v8bf hi = *(const v8bf*)(p + 16);
  v16bf r;
#pragma unroll
  for (int i = 0; i < 8; ++i) { r[i] = lo[i]; r[i + 8] = hi[i]; }
  return r;
}

// Same A-layout gather straight from a row-major f32 global matrix (pitch 64),
// converting to bf16 in flight (used for the q readout: q is read exactly once).
__device__ __forceinline__ v16bf load_tile_gf32(const float* src, int rowBase,
                                                int pitch, int k0, int lane) {
  const int half = lane >> 4, l = lane & 15;
  const float* p = src + (size_t)(rowBase + l) * pitch + k0 + half * 8;
  v16bf r;
#pragma unroll
  for (int i = 0; i < 8; ++i) {
    r[i]     = (bf16_t)p[i];
    r[i + 8] = (bf16_t)p[i + 16];
  }
  return r;
}

__device__ __forceinline__ v8f wmma_bf16(v16bf a, v16bf b, v8f c) {
  // v_wmma_f32_16x16x32_bf16  (neg_a, A, neg_b, B, c_mod, C, reuse_a, reuse_b)
  return __builtin_amdgcn_wmma_f32_16x16x32_bf16(false, a, false, b,
                                                 (short)0, c, false, false);
}

__global__ __launch_bounds__(256, 1)
void iplr_scan_kernel(const float* __restrict__ q, const float* __restrict__ k,
                      const float* __restrict__ v, const float* __restrict__ lr,
                      const float* __restrict__ state0, float* __restrict__ y) {
  __shared__ __align__(16) bf16_t sA[CHUNKL * PA];   // a = -k (chunk)   36864B
  __shared__ __align__(16) bf16_t sST[NDIM * PS];    // S^T bf16          9216B
  __shared__ __align__(16) bf16_t sBT[NDIM * PB];    // b^T  (block, Kpad) 4096B
  __shared__ __align__(16) bf16_t sWT[NDIM * PB];    // w^T  (block, Kpad) 4096B
  __shared__ __align__(16) bf16_t sBb[16 * PA];      // b    (block, rows) 2304B
  __shared__ __align__(16) float  sT[16 * PT];       // t = v + a@S        4160B
  __shared__ __align__(16) float  sAB[16 * PAB];     // a_i b_i^T tile     1088B
  __shared__ __align__(16) float  sLR[CHUNKL];       // chunk lr           1024B

  const int bh   = blockIdx.x;            // b*H + h, 64 sequences
  const int tid  = threadIdx.x;
  const int wave = tid >> 5;
  const int lane = tid & 31;
  const int half = lane >> 4;
  const int l    = lane & 15;

  const size_t base  = (size_t)bh * TLEN * NDIM;
  const size_t lrb   = (size_t)bh * TLEN;
  const size_t sbase = (size_t)bh * NDIM * NDIM;

  // ---- persistent state: wave owns 2 of the 16 (p0,q0) 16x16 tiles of S ----
  v8f sreg[2];
  int p0[2], q0[2];
#pragma unroll
  for (int ti = 0; ti < 2; ++ti) {
    const int tile = wave * 2 + ti;
    p0[ti] = (tile >> 2) * 16;            // state row (k-feature) block
    q0[ti] = (tile & 3) * 16;             // state col (v-feature) block
    v8f c;
#pragma unroll
    for (int g = 0; g < 8; ++g)           // C layout: m = half*8+g, n = l
      c[g] = state0[sbase + (size_t)(p0[ti] + half * 8 + g) * NDIM + q0[ti] + l];
    sreg[ti] = c;
    v8bf sb;
#pragma unroll
    for (int g = 0; g < 8; ++g) sb[g] = (bf16_t)c[g];
    *(v8bf*)&sST[(q0[ti] + l) * PS + p0[ti] + half * 8] = sb;
  }
  // zero K-pad halves of b^T / w^T once (rewritten valid halves each block)
  for (int i = tid; i < NDIM * PB; i += 256) { sBT[i] = (bf16_t)0.f; sWT[i] = (bf16_t)0.f; }

  for (int ch = 0; ch < NCHUNK; ++ch) {
    __syncthreads();
    const size_t cbase = base + (size_t)ch * CHUNKL * NDIM;

    // ---- stage chunk: a = -k into LDS (bf16), lr into LDS ------------------
    sLR[tid] = lr[lrb + ch * CHUNKL + tid];
    const float4* k4 = (const float4*)(k + cbase);
#pragma unroll
    for (int it = 0; it < 16; ++it) {     // 4096 float4 / 256 threads
      const int e   = tid + it * 256;
      const int row = e >> 4;
      const int c4  = (e & 15) << 2;
      const float4 kv = k4[e];
      v4bf a4;
      a4[0] = (bf16_t)(-kv.x); a4[1] = (bf16_t)(-kv.y);
      a4[2] = (bf16_t)(-kv.z); a4[3] = (bf16_t)(-kv.w);
      *(v4bf*)&sA[row * PA + c4] = a4;
    }
    __syncthreads();

    // ---- block-16 recurrence over the chunk --------------------------------
    for (int blk = 0; blk < 16; ++blk) {
      // b = k*lr = -a*lr : fill b^T [p][r] (K-padded) and b row-major [r][p]
#pragma unroll
      for (int it = 0; it < 4; ++it) {
        const int i = tid + it * 256;     // 1024 entries
        const int p = i >> 4, r = i & 15;
        const int rg = blk * 16 + r;
        const bf16_t bb = (bf16_t)(-(float)sA[rg * PA + p] * sLR[rg]);
        sBT[p * PB + r]  = bb;
        sBb[r * PA + p]  = bb;
      }
      __syncthreads();

      if (wave < 4) {                     // t = v_blk + a_blk @ S  (4 N-tiles)
        const int qq = wave * 16;
        v8f c;
#pragma unroll
        for (int g = 0; g < 8; ++g)       // C init straight from global v
          c[g] = v[cbase + (size_t)(blk * 16 + half * 8 + g) * NDIM + qq + l];
        c = wmma_bf16(load_tile_lds(sA, blk * 16, PA, 0,  lane),
                      load_tile_lds(sST, qq, PS, 0,  lane), c);
        c = wmma_bf16(load_tile_lds(sA, blk * 16, PA, 32, lane),
                      load_tile_lds(sST, qq, PS, 32, lane), c);
#pragma unroll
        for (int g = 0; g < 8; ++g)
          sT[(half * 8 + g) * PT + qq + l] = c[g];
      } else if (wave == 4) {             // AB = a_blk @ b_blk^T (one tile)
        v8f c = {};
        c = wmma_bf16(load_tile_lds(sA, blk * 16, PA, 0,  lane),
                      load_tile_lds(sBb, 0, PA, 0,  lane), c);
        c = wmma_bf16(load_tile_lds(sA, blk * 16, PA, 32, lane),
                      load_tile_lds(sBb, 0, PA, 32, lane), c);
#pragma unroll
        for (int g = 0; g < 8; ++g)
          sAB[(half * 8 + g) * PAB + l] = c[g];
      }
      __syncthreads();

      // 16x16 unit-lower substitution: w = t + tril(AB,-1) @ w, one column/lane
      if (tid < 64) {
        float w[16];
        w[0] = sT[tid];
#pragma unroll
        for (int r = 1; r < 16; ++r) {
          float acc = sT[r * PT + tid];
          for (int j = 0; j < r; ++j) acc += sAB[r * PAB + j] * w[j];
          w[r] = acc;
        }
#pragma unroll
        for (int r = 0; r < 16; ++r) sWT[tid * PB + r] = (bf16_t)w[r];
      }
      __syncthreads();

      // S += b_blk^T @ w_blk  (K=16 padded to 32); refresh bf16 S^T mirror
#pragma unroll
      for (int ti = 0; ti < 2; ++ti) {
        sreg[ti] = wmma_bf16(load_tile_lds(sBT, p0[ti], PB, 0, lane),
                             load_tile_lds(sWT, q0[ti], PB, 0, lane), sreg[ti]);
        v8bf sb;
#pragma unroll
        for (int g = 0; g < 8; ++g) sb[g] = (bf16_t)sreg[ti][g];
        *(v8bf*)&sST[(q0[ti] + l) * PS + p0[ti] + half * 8] = sb;
      }
      __syncthreads();
    }

    // ---- prefetch next chunk's k and q while the readout GEMM runs ---------
    if (ch + 1 < NCHUNK) {
      const float* nk = k + cbase + CHUNKL * NDIM;
      const float* nq = q + cbase + CHUNKL * NDIM;
      __builtin_prefetch(nk + (size_t)tid * 64, 0, 0);   // 256B stride covers chunk
      __builtin_prefetch(nq + (size_t)tid * 64, 0, 0);
    }

    // ---- readout: y_chunk = q_chunk @ S_end (64 tiles, 8 per wave) ---------
#pragma unroll
    for (int ri = 0; ri < 2; ++ri) {
      const int rb = wave * 2 + ri;
      const v16bf a0 = load_tile_gf32(q + cbase, rb * 16, NDIM, 0,  lane);
      const v16bf a1 = load_tile_gf32(q + cbase, rb * 16, NDIM, 32, lane);
#pragma unroll
      for (int qt = 0; qt < 4; ++qt) {
        const int qq = qt * 16;
        v8f c = {};
        c = wmma_bf16(a0, load_tile_lds(sST, qq, PS, 0,  lane), c);
        c = wmma_bf16(a1, load_tile_lds(sST, qq, PS, 32, lane), c);
        float* yp = y + cbase + (size_t)(rb * 16) * NDIM;
#pragma unroll
        for (int g = 0; g < 8; ++g)
          yp[(size_t)(half * 8 + g) * NDIM + qq + l] = c[g];
      }
    }
  }
}

extern "C" void kernel_launch(void* const* d_in, const int* in_sizes, int n_in,
                              void* d_out, int out_size, void* d_ws, size_t ws_size,
                              hipStream_t stream) {
  const float* q  = (const float*)d_in[0];
  const float* k  = (const float*)d_in[1];
  const float* v  = (const float*)d_in[2];
  const float* lr = (const float*)d_in[3];
  const float* s0 = (const float*)d_in[4];
  float* y = (float*)d_out;
  (void)in_sizes; (void)n_in; (void)out_size; (void)d_ws; (void)ws_size;
  // One workgroup per (b,h) sequence: B*H = 64 blocks, 8 waves each.
  iplr_scan_kernel<<<dim3(64), dim3(256), 0, stream>>>(q, k, v, lr, s0, y);
}
